// VL_Align_71665824301089
// MI455X (gfx1250) — matrix operand order repaired and checked by reading, not compiled, not run on hardware
//
#include <hip/hip_runtime.h>
#include <cstdint>
#include <cstddef>

#define BS   64
#define NQ   900
#define LSEQ 1024
#define LANG 768
#define HID  256
#define NEXP 8

typedef __attribute__((ext_vector_type(16))) __bf16          v16bf;
typedef __attribute__((ext_vector_type(8)))  float           v8f;
typedef __attribute__((ext_vector_type(16))) unsigned short  v16u;
typedef __attribute__((ext_vector_type(8)))  unsigned short  v8u;

static __device__ __forceinline__ unsigned short f2bf(float f) {
    unsigned int u = __float_as_uint(f);
    u += 0x7fffu + ((u >> 16) & 1u);   // round-to-nearest-even
    return (unsigned short)(u >> 16);
}

static __device__ __forceinline__ float block_reduce(float v, float* s) {
    int t = threadIdx.x;
    s[t] = v;
    __syncthreads();
    for (int off = 128; off > 0; off >>= 1) {
        if (t < off) s[t] += s[t + off];
        __syncthreads();
    }
    float r = s[0];
    __syncthreads();
    return r;
}

// Assemble an A fragment (contiguous K order: 16 consecutive K per lane half, k = 16h + s).
static __device__ __forceinline__ v16bf load_a_frag(const unsigned short* row, int k0, int ka0) {
    v8u lo = *(const v8u*)(row + k0 + 2 * ka0);
    v8u hi = *(const v8u*)(row + k0 + 2 * ka0 + 8);
    v16u au;
    #pragma unroll
    for (int i = 0; i < 8; i++) { au[i] = lo[i]; au[i + 8] = hi[i]; }
    return __builtin_bit_cast(v16bf, au);
}

// ---------------- prep: x -> bf16 ----------------
__global__ void __launch_bounds__(256) cvt_x_kernel(const float* __restrict__ x,
                                                    unsigned short* __restrict__ xb, int n) {
    int i = blockIdx.x * 256 + threadIdx.x;
    if (i < n) xb[i] = f2bf(x[i]);
}

// ---------------- prep: weights -> bf16, transposed to [w][n][k] ----------------
__global__ void __launch_bounds__(256) cvt_w_kernel(const float* __restrict__ proj_w,
                                                    const float* __restrict__ expert_w,
                                                    unsigned short* __restrict__ wts) {
    int i = blockIdx.x * 256 + threadIdx.x;
    if (i >= 9 * HID * LANG) return;
    int w   = i / (HID * LANG);
    int rem = i % (HID * LANG);
    int n   = rem / LANG;
    int k   = rem % LANG;
    float s = (w == 0) ? proj_w[k * HID + n]
                       : expert_w[(((size_t)(w - 1)) * LANG + k) * HID + n];
    wts[i] = f2bf(s);   // dst index == ((w*HID+n)*LANG + k)
}

// ---------------- normalize + gate softmax/top2 + token bias ----------------
__global__ void __launch_bounds__(256) norm_gate_kernel(const float* __restrict__ emb,
                                                        const float* __restrict__ gate_w,
                                                        const float* __restrict__ gate_b,
                                                        const float* __restrict__ bias_lang,
                                                        const float* __restrict__ bias0,
                                                        unsigned short* __restrict__ embb,
                                                        float* __restrict__ combine,
                                                        float* __restrict__ tbias) {
    __shared__ float sred[256];
    int b = blockIdx.y, l = blockIdx.x, t = threadIdx.x;
    const float* row = emb + ((size_t)(b * LSEQ + l)) * LANG;
    float v0 = row[t], v1 = row[t + 256], v2 = row[t + 512];

    float ss = block_reduce(v0 * v0 + v1 * v1 + v2 * v2, sred);
    float rnorm = rsqrtf(ss);

    unsigned short* orow = embb + ((size_t)(b * LSEQ + l)) * LANG;
    orow[t]       = f2bf(v0 * rnorm);
    orow[t + 256] = f2bf(v1 * rnorm);
    orow[t + 512] = f2bf(v2 * rnorm);

    float bd = v0 * bias_lang[t] + v1 * bias_lang[t + 256] + v2 * bias_lang[t + 512];
    bd = block_reduce(bd, sred);

    float g[NEXP];
    #pragma unroll
    for (int e = 0; e < NEXP; e++) {
        float p = v0 * gate_w[t * NEXP + e]
                + v1 * gate_w[(t + 256) * NEXP + e]
                + v2 * gate_w[(t + 512) * NEXP + e];
        g[e] = block_reduce(p, sred);
    }

    if (t == 0) {
        float logits[NEXP], p[NEXP];
        float m = -1e30f;
        #pragma unroll
        for (int e = 0; e < NEXP; e++) {
            logits[e] = g[e] * rnorm + gate_b[e];
            m = fmaxf(m, logits[e]);
        }
        float s = 0.f;
        #pragma unroll
        for (int e = 0; e < NEXP; e++) { p[e] = __expf(logits[e] - m); s += p[e]; }
        float inv = 1.f / s;
        int i1 = 0;
        for (int e = 1; e < NEXP; e++) if (p[e] > p[i1]) i1 = e;
        int i2 = -1;
        for (int e = 0; e < NEXP; e++) {
            if (e == i1) continue;
            if (i2 < 0 || p[e] > p[i2]) i2 = e;
        }
        float* crow = combine + ((size_t)(b * LSEQ + l)) * NEXP;
        #pragma unroll
        for (int e = 0; e < NEXP; e++)
            crow[e] = (e == i1 || e == i2) ? p[e] * inv : 0.f;
        tbias[b * LSEQ + l] = bd * rnorm + bias0[0];
    }
}

// ---------------- tokens = 0.5*proj + 0.5*MoE  (9 WMMA accumulators / A-tile) ----------------
__global__ void __launch_bounds__(256) tokens_moe_kernel(const unsigned short* __restrict__ embb,
                                                         const unsigned short* __restrict__ wts,
                                                         const float* __restrict__ proj_b,
                                                         const float* __restrict__ expert_b,
                                                         const float* __restrict__ combine,
                                                         unsigned short* __restrict__ tok) {
    int b    = blockIdx.z;
    int l0   = blockIdx.y * 16;
    int wave = threadIdx.x >> 5;
    int lane = threadIdx.x & 31;
    int n0   = blockIdx.x * 128 + wave * 16;
    int N    = lane & 15;
    int half = lane >> 4;

    const unsigned short* arow = embb + ((size_t)(b * LSEQ + l0 + N)) * LANG;
    int ka0 = half * 8;      // A K offset / 2 (load_a_frag reads k = 16h + s)
    int kb0 = half * 16;     // B K-offset, contiguous k = 16h + s as for A

    v8f acc[9];
    #pragma unroll
    for (int w = 0; w < 9; w++) acc[w] = (v8f){0.f, 0.f, 0.f, 0.f, 0.f, 0.f, 0.f, 0.f};

    for (int k0 = 0; k0 < LANG; k0 += 32) {
        v16bf a = load_a_frag(arow, k0, ka0);
        #pragma unroll
        for (int w = 0; w < 9; w++) {
            const unsigned short* bp = wts + ((size_t)w * HID + n0 + N) * LANG + k0 + kb0;
            v16bf bm = __builtin_bit_cast(v16bf, *(const v16u*)bp);
            acc[w] = __builtin_amdgcn_wmma_f32_16x16x32_bf16(
                false, a, false, bm, (short)0, acc[w], false, false);
        }
    }

    int n = n0 + N;
    float pb = proj_b[n];
    #pragma unroll
    for (int r = 0; r < 8; r++) {
        int M = r + half * 8;
        int l = l0 + M;
        const float* crow = combine + ((size_t)(b * LSEQ + l)) * NEXP;
        float moe = 0.f;
        #pragma unroll
        for (int e = 0; e < NEXP; e++)
            moe += crow[e] * (acc[1 + e][r] + expert_b[e * HID + n]);
        float val = 0.5f * (acc[0][r] + pb) + 0.5f * moe;
        tok[((size_t)(b * LSEQ + l)) * HID + n] = f2bf(val);
    }
}

// ---------------- logit = clip(x·tokens^T * inv_scale + bias), 2x2 register tiling ----------------
__global__ void __launch_bounds__(256) logits_kernel(const unsigned short* __restrict__ xb,
                                                     const unsigned short* __restrict__ tok,
                                                     const float* __restrict__ tbias,
                                                     const float* __restrict__ log_scale,
                                                     float* __restrict__ out) {
    int b    = blockIdx.z;
    int q0   = blockIdx.y * 32;                 // 2 M tiles per wave
    int wave = threadIdx.x >> 5;
    int lane = threadIdx.x & 31;
    int l0   = blockIdx.x * 256 + wave * 32;    // 2 N tiles per wave
    int N    = lane & 15;
    int half = lane >> 4;

    int qa0 = q0 + N;      if (qa0 > NQ - 1) qa0 = NQ - 1;  // clamp loads; masked at store
    int qa1 = q0 + 16 + N; if (qa1 > NQ - 1) qa1 = NQ - 1;
    const unsigned short* arow0 = xb  + ((size_t)(b * NQ + qa0)) * HID;
    const unsigned short* arow1 = xb  + ((size_t)(b * NQ + qa1)) * HID;
    const unsigned short* brow0 = tok + ((size_t)(b * LSEQ + l0 + N)) * HID;
    const unsigned short* brow1 = tok + ((size_t)(b * LSEQ + l0 + 16 + N)) * HID;
    int ka0 = half * 8;
    int kb0 = half * 16;

    v8f acc00 = (v8f){0.f, 0.f, 0.f, 0.f, 0.f, 0.f, 0.f, 0.f};
    v8f acc01 = acc00, acc10 = acc00, acc11 = acc00;

    #pragma unroll
    for (int k0 = 0; k0 < HID; k0 += 32) {
        v16bf a0 = load_a_frag(arow0, k0, ka0);
        v16bf a1 = load_a_frag(arow1, k0, ka0);
        v16bf b0 = __builtin_bit_cast(v16bf, *(const v16u*)(brow0 + k0 + kb0));
        v16bf b1 = __builtin_bit_cast(v16bf, *(const v16u*)(brow1 + k0 + kb0));
        acc00 = __builtin_amdgcn_wmma_f32_16x16x32_bf16(false, a0, false, b0, (short)0, acc00, false, false);
        acc01 = __builtin_amdgcn_wmma_f32_16x16x32_bf16(false, a0, false, b1, (short)0, acc01, false, false);
        acc10 = __builtin_amdgcn_wmma_f32_16x16x32_bf16(false, a1, false, b0, (short)0, acc10, false, false);
        acc11 = __builtin_amdgcn_wmma_f32_16x16x32_bf16(false, a1, false, b1, (short)0, acc11, false, false);
    }

    float inv = __expf(-log_scale[0]);
    int lA = l0 + N;
    int lB = l0 + 16 + N;
    float tb0 = tbias[b * LSEQ + lA];
    float tb1 = tbias[b * LSEQ + lB];

    #pragma unroll
    for (int r = 0; r < 8; r++) {
        int qA = q0 + r + half * 8;        // M tile 0
        int qB = qA + 16;                  // M tile 1
        if (qA < NQ) {
            float v = fminf(fmaxf(acc00[r] * inv + tb0, -50000.f), 50000.f);
            out[((size_t)(b * NQ + qA)) * LSEQ + lA] = v;
            v = fminf(fmaxf(acc01[r] * inv + tb1, -50000.f), 50000.f);
            out[((size_t)(b * NQ + qA)) * LSEQ + lB] = v;
        }
        if (qB < NQ) {
            float v = fminf(fmaxf(acc10[r] * inv + tb0, -50000.f), 50000.f);
            out[((size_t)(b * NQ + qB)) * LSEQ + lA] = v;
            v = fminf(fmaxf(acc11[r] * inv + tb1, -50000.f), 50000.f);
            out[((size_t)(b * NQ + qB)) * LSEQ + lB] = v;
        }
    }
}

extern "C" void kernel_launch(void* const* d_in, const int* in_sizes, int n_in,
                              void* d_out, int out_size, void* d_ws, size_t ws_size,
                              hipStream_t stream) {
    const float* x         = (const float*)d_in[0];
    const float* embedding = (const float*)d_in[1];
    const float* proj_w    = (const float*)d_in[2];
    const float* proj_b    = (const float*)d_in[3];
    const float* gate_w    = (const float*)d_in[4];
    const float* gate_b    = (const float*)d_in[5];
    const float* expert_w  = (const float*)d_in[6];
    const float* expert_b  = (const float*)d_in[7];
    const float* bias_lang = (const float*)d_in[8];
    const float* bias0     = (const float*)d_in[9];
    const float* log_scale = (const float*)d_in[10];
    float* out = (float*)d_out;

    char* ws = (char*)d_ws;
    size_t off = 0;
    auto alloc = [&](size_t bytes) -> void* {
        void* p = ws + off;
        off = (off + bytes + 255) & ~(size_t)255;
        return p;
    };
    unsigned short* embb    = (unsigned short*)alloc((size_t)BS * LSEQ * LANG * 2);
    unsigned short* xb      = (unsigned short*)alloc((size_t)BS * NQ * HID * 2);
    unsigned short* tok     = (unsigned short*)alloc((size_t)BS * LSEQ * HID * 2);
    unsigned short* wts     = (unsigned short*)alloc((size_t)9 * HID * LANG * 2);
    float*          combine = (float*)alloc((size_t)BS * LSEQ * NEXP * 4);
    float*          tbias   = (float*)alloc((size_t)BS * LSEQ * 4);
    (void)ws_size; (void)in_sizes; (void)n_in; (void)out_size;

    {
        int n = BS * NQ * HID;
        cvt_x_kernel<<<(n + 255) / 256, 256, 0, stream>>>(x, xb, n);
    }
    {
        int n = 9 * HID * LANG;
        cvt_w_kernel<<<(n + 255) / 256, 256, 0, stream>>>(proj_w, expert_w, wts);
    }
    norm_gate_kernel<<<dim3(LSEQ, BS), 256, 0, stream>>>(
        embedding, gate_w, gate_b, bias_lang, bias0, embb, combine, tbias);
    tokens_moe_kernel<<<dim3(HID / 128, LSEQ / 16, BS), 256, 0, stream>>>(
        embb, wts, proj_b, expert_b, combine, tok);
    logits_kernel<<<dim3(LSEQ / 256, (NQ + 31) / 32, BS), 256, 0, stream>>>(
        xb, tok, tbias, log_scale, out);
}
